// Attention_7765300871328
// MI455X (gfx1250) — matrix operand order repaired and checked by reading, not compiled, not run on hardware
//
#include <hip/hip_runtime.h>

typedef __attribute__((ext_vector_type(16))) _Float16 v16h;
typedef __attribute__((ext_vector_type(8)))  float    v8f;

#define T_      2048
#define HID     2048
#define HQ_     32
#define G_      2
#define D_      64
#define NCHUNK  127
#define WINDOW_ 512
#define TOPK_   16

#if defined(__has_builtin)
#if __has_builtin(__builtin_amdgcn_tensor_load_to_lds) && __has_builtin(__builtin_amdgcn_s_wait_tensorcnt)
#define USE_TDM 1
#endif
#endif

// ---------------- WMMA fragment helpers (gfx1250 lane layouts) ----------------

static __device__ __forceinline__ v16h load_a_frag(const _Float16* base, int ld, int lane) {
  // 16x32 f16 A: lanes 0-15 -> M=lane, K={0..7,16..23}; lanes 16-31 -> M=lane-16, K={8..15,24..31}
  union { v16h v; unsigned u[8]; } r;
  const _Float16* p = base + (size_t)(lane & 15) * ld + ((lane >> 4) << 3);
  #pragma unroll
  for (int j = 0; j < 4; ++j) r.u[j]     = *(const unsigned*)(p + 2 * j);
  #pragma unroll
  for (int j = 0; j < 4; ++j) r.u[4 + j] = *(const unsigned*)(p + 16 + 2 * j);
  return r.v;
}

static __device__ __forceinline__ v16h load_b_frag(const _Float16* base, int ld, int lane) {
  // 32x16 f16 B from row-major [N][K]: lanes 0-15 -> N=lane, K={0..7,16..23}; lanes 16-31 -> N=lane-16, K={8..15,24..31}
  union { v16h v; unsigned u[8]; } r;
  const _Float16* p = base + (size_t)(lane & 15) * ld + ((lane >> 4) << 3);
  #pragma unroll
  for (int j = 0; j < 4; ++j) r.u[j]     = *(const unsigned*)(p + 2 * j);
  #pragma unroll
  for (int j = 0; j < 4; ++j) r.u[4 + j] = *(const unsigned*)(p + 16 + 2 * j);
  return r.v;
}

static __device__ __forceinline__ v8f wmma32(v16h a, v16h b, v8f c) {
  return __builtin_amdgcn_wmma_f32_16x16x32_f16(false, a, false, b, (short)0, c, false, false);
}

// ---------------- Tensor Data Mover: 2D f16 tile -> LDS ----------------

#ifdef USE_TDM
typedef __attribute__((ext_vector_type(4))) unsigned tdm_v4u;
typedef __attribute__((ext_vector_type(8))) int      tdm_v8i;
typedef __attribute__((ext_vector_type(4))) int      tdm_v4i;

// tile_x elements along contiguous dim0, tile_y rows, row stride = stride_x elements (f16, 2B).
static __device__ __forceinline__ void tdm_load_tile_f16(unsigned lds_byte_off, const void* gptr,
                                                         unsigned tile_x, unsigned tile_y,
                                                         unsigned stride_x) {
  unsigned long long ga = (unsigned long long)(size_t)gptr;
  tdm_v4u g0;
  g0[0] = 1u;                                                 // count=1 (valid user descriptor)
  g0[1] = lds_byte_off;                                       // D#.lds_addr
  g0[2] = (unsigned)(ga & 0xFFFFFFFFull);                     // D#.global_addr[31:0]
  g0[3] = (unsigned)((ga >> 32) & 0x01FFFFFFull) | (2u << 30); // global_addr[56:32] | type=2
  tdm_v8i g1;
  g1[0] = (int)(1u << 16);                                    // workgroup_mask=0, data_size=1 (2B)
  g1[1] = (int)((stride_x & 0xFFFFu) << 16);                  // tensor_dim0[15:0] (use full row len)
  g1[2] = (int)((stride_x >> 16) | ((tile_y & 0xFFFFu) << 16)); // tensor_dim0 hi | tensor_dim1 lo
  g1[3] = (int)((tile_y >> 16) | (tile_x << 16));             // tensor_dim1 hi | tile_dim0
  g1[4] = (int)(tile_y & 0xFFFFu);                            // tile_dim1 (tile_dim2 = 0)
  g1[5] = (int)stride_x;                                      // tensor_dim0_stride[31:0]
  g1[6] = 0;                                                  // stride hi | tensor_dim1_stride lo
  g1[7] = 0;
  tdm_v4i z4 = {};
  tdm_v8i z8 = {};
  // clang-23 / amdgpu-toolchain arity: (g0, g1, g2, g3, g4, cpol)
  __builtin_amdgcn_tensor_load_to_lds(g0, g1, z4, z4, z8, 0);
}

static __device__ __forceinline__ unsigned lds_off_of(const void* p) {
  return (unsigned)(unsigned long long)(size_t)p;   // LDS aperture: low 32 bits = LDS offset
}
#endif

// ---------------- generic f32->f16 convert ----------------

__global__ void f32_to_f16_kernel(const float* __restrict__ src, _Float16* __restrict__ dst, size_t n) {
  size_t stride = (size_t)gridDim.x * blockDim.x;
  for (size_t i = (size_t)blockIdx.x * blockDim.x + threadIdx.x; i < n; i += stride)
    dst[i] = (_Float16)src[i];
}

// ---------------- tiled WMMA GEMM: C[M,N] = A[M,K] * B[N,K]^T, f32 out ----------------
// BM=BN=128, BK=32, 8 waves: 4(m) x 2(n), each wave 32x64 = 2x4 WMMA tiles.
// Double-buffered LDS tiles staged by the Tensor Data Mover (TENSORcnt) when available.

__global__ __launch_bounds__(256) void gemm_f16(const _Float16* __restrict__ A,
                                                const _Float16* __restrict__ B,
                                                float* __restrict__ C,
                                                int K, int ldc,
                                                size_t strA, size_t strB, size_t strC) {
  __shared__ _Float16 As[2][128 * 32];
  __shared__ _Float16 Bs[2][128 * 32];
  A += (size_t)blockIdx.z * strA;
  B += (size_t)blockIdx.z * strB;
  C += (size_t)blockIdx.z * strC;
  const int tid  = threadIdx.x;
  const int lane = tid & 31;
  const int wave = tid >> 5;
  const int wm   = wave >> 1, wn = wave & 1;
  const int m0 = blockIdx.y * 128, n0 = blockIdx.x * 128;
  v8f zacc = {};
  v8f acc[2][4];
  #pragma unroll
  for (int i = 0; i < 2; ++i)
    #pragma unroll
    for (int j = 0; j < 4; ++j) acc[i][j] = zacc;

  const int nk = K >> 5;

#ifdef USE_TDM
  // wave 0 drives the TDM pipeline; TENSORcnt tracks completion.
  if (wave == 0) {
    tdm_load_tile_f16(lds_off_of(&As[0][0]), A + (size_t)m0 * K, 32u, 128u, (unsigned)K);
    tdm_load_tile_f16(lds_off_of(&Bs[0][0]), B + (size_t)n0 * K, 32u, 128u, (unsigned)K);
  }
  for (int i = 0; i < nk; ++i) {
    const int buf = i & 1;
    if (wave == 0) {
      if (i + 1 < nk) {
        const int kk = (i + 1) << 5;
        tdm_load_tile_f16(lds_off_of(&As[buf ^ 1][0]), A + (size_t)m0 * K + kk, 32u, 128u, (unsigned)K);
        tdm_load_tile_f16(lds_off_of(&Bs[buf ^ 1][0]), B + (size_t)n0 * K + kk, 32u, 128u, (unsigned)K);
        __builtin_amdgcn_s_wait_tensorcnt(2);   // tile i landed; tile i+1 still in flight
      } else {
        __builtin_amdgcn_s_wait_tensorcnt(0);
      }
    }
    __syncthreads();
    v16h af[2], bf[4];
    #pragma unroll
    for (int mi = 0; mi < 2; ++mi) af[mi] = load_a_frag(&As[buf][0] + (wm * 32 + mi * 16) * 32, 32, lane);
    #pragma unroll
    for (int ni = 0; ni < 4; ++ni) bf[ni] = load_b_frag(&Bs[buf][0] + (wn * 64 + ni * 16) * 32, 32, lane);
    #pragma unroll
    for (int mi = 0; mi < 2; ++mi)
      #pragma unroll
      for (int ni = 0; ni < 4; ++ni)
        acc[mi][ni] = wmma32(af[mi], bf[ni], acc[mi][ni]);
    __syncthreads();
  }
#else
  for (int i = 0; i < nk; ++i) {
    const int kk = i << 5;
    const int buf = i & 1;
    for (int j = tid; j < 512; j += 256) {          // 128x32 f16 = 512 uint4 chunks per tile
      int row = j >> 2, seg = j & 3;
      *(uint4*)(&As[buf][0] + row * 32 + seg * 8) = *(const uint4*)(A + (size_t)(m0 + row) * K + kk + seg * 8);
      *(uint4*)(&Bs[buf][0] + row * 32 + seg * 8) = *(const uint4*)(B + (size_t)(n0 + row) * K + kk + seg * 8);
      if (kk + 32 < K) {
        __builtin_prefetch(A + (size_t)(m0 + row) * K + kk + 32 + seg * 8, 0, 1);
        __builtin_prefetch(B + (size_t)(n0 + row) * K + kk + 32 + seg * 8, 0, 1);
      }
    }
    __syncthreads();
    v16h af[2], bf[4];
    #pragma unroll
    for (int mi = 0; mi < 2; ++mi) af[mi] = load_a_frag(&As[buf][0] + (wm * 32 + mi * 16) * 32, 32, lane);
    #pragma unroll
    for (int ni = 0; ni < 4; ++ni) bf[ni] = load_b_frag(&Bs[buf][0] + (wn * 64 + ni * 16) * 32, 32, lane);
    #pragma unroll
    for (int mi = 0; mi < 2; ++mi)
      #pragma unroll
      for (int ni = 0; ni < 4; ++ni)
        acc[mi][ni] = wmma32(af[mi], bf[ni], acc[mi][ni]);
    __syncthreads();
  }
#endif

  const int mo = (lane >> 4) * 8, nn = lane & 15;
  #pragma unroll
  for (int mi = 0; mi < 2; ++mi)
    #pragma unroll
    for (int ni = 0; ni < 4; ++ni) {
      float* cp = C + (size_t)(m0 + wm * 32 + mi * 16 + mo) * ldc + n0 + wn * 64 + ni * 16 + nn;
      #pragma unroll
      for (int r = 0; r < 8; ++r) cp[(size_t)r * ldc] = acc[mi][ni][r];
    }
}

// ---------------- gate: sigmoid(x @ Wg^T), tiny ----------------

__global__ __launch_bounds__(256) void gate_kernel(const float* __restrict__ x,
                                                   const float* __restrict__ Wg,
                                                   float* __restrict__ gate) {
  __shared__ float red[3][256];
  int t = blockIdx.x, tid = threadIdx.x;
  float s0 = 0.f, s1 = 0.f, s2 = 0.f;
  for (int k = tid; k < HID; k += 256) {
    float xv = x[(size_t)t * HID + k];
    s0 += xv * Wg[k]; s1 += xv * Wg[HID + k]; s2 += xv * Wg[2 * HID + k];
  }
  red[0][tid] = s0; red[1][tid] = s1; red[2][tid] = s2;
  __syncthreads();
  for (int off = 128; off > 0; off >>= 1) {
    if (tid < off) {
      red[0][tid] += red[0][tid + off];
      red[1][tid] += red[1][tid + off];
      red[2][tid] += red[2][tid + off];
    }
    __syncthreads();
  }
  if (tid < 3) gate[t * 3 + tid] = 1.f / (1.f + __expf(-red[tid][0]));
}

// ---------------- RoPE + repack into WMMA-friendly f16 layouts ----------------
// qab[g][(t*16+h)][d], kb[t][g][d], vb[t][g][d]

__global__ __launch_bounds__(256) void rope_pack(const float* __restrict__ qf,
                                                 const float* __restrict__ kf,
                                                 const float* __restrict__ vf,
                                                 _Float16* __restrict__ qab,
                                                 _Float16* __restrict__ kb,
                                                 _Float16* __restrict__ vb) {
  const int t = blockIdx.x, tid = threadIdx.x;
  for (int i = tid; i < HQ_ * 32; i += 256) {
    int h = i >> 5, j = i & 31;
    float inv = powf(10000.f, -(float)j / 32.f);
    float fr = (float)t * inv;
    float c = cosf(fr), s = sinf(fr);
    float x1 = qf[((size_t)t * HQ_ + h) * D_ + j];
    float x2 = qf[((size_t)t * HQ_ + h) * D_ + 32 + j];
    int g = h >> 4, hh = h & 15;
    size_t base = ((size_t)g * (T_ * 16) + (size_t)t * 16 + hh) * D_;
    qab[base + j]      = (_Float16)(x1 * c - x2 * s);
    qab[base + 32 + j] = (_Float16)(x2 * c + x1 * s);
  }
  for (int i = tid; i < G_ * 32; i += 256) {
    int g = i >> 5, j = i & 31;
    float inv = powf(10000.f, -(float)j / 32.f);
    float fr = (float)t * inv;
    float c = cosf(fr), s = sinf(fr);
    float x1 = kf[((size_t)t * G_ + g) * D_ + j];
    float x2 = kf[((size_t)t * G_ + g) * D_ + 32 + j];
    size_t base = ((size_t)t * G_ + g) * D_;
    kb[base + j]      = (_Float16)(x1 * c - x2 * s);
    kb[base + 32 + j] = (_Float16)(x2 * c + x1 * s);
  }
  for (int i = tid; i < G_ * D_; i += 256)
    vb[(size_t)t * G_ * D_ + i] = (_Float16)vf[(size_t)t * G_ * D_ + i];
}

// ---------------- build compression windows kw/vw [G][128][2048] (row 127 = pad 0) ----------------

__global__ __launch_bounds__(256) void build_windows(const _Float16* __restrict__ kb,
                                                     const _Float16* __restrict__ vb,
                                                     _Float16* __restrict__ kwb,
                                                     _Float16* __restrict__ vwb) {
  int c = blockIdx.x, g = blockIdx.y;
  for (int f = threadIdx.x; f < 2048; f += 256) {
    _Float16 a = (_Float16)0.f, b = (_Float16)0.f;
    if (c < NCHUNK) {
      int kp = f >> 6, d = f & 63;
      int t = c * 16 + kp;
      a = kb[((size_t)t * G_ + g) * D_ + d];
      b = vb[((size_t)t * G_ + g) * D_ + d];
    }
    kwb[((size_t)g * 128 + c) * 2048 + f] = a;
    vwb[((size_t)g * 128 + c) * 2048 + f] = b;
  }
}

// ---------------- transpose compress weights to [G][128(dpad)][2048] f16 ----------------

__global__ __launch_bounds__(256) void prep_cw(const float* __restrict__ CKW,
                                               const float* __restrict__ CVW,
                                               _Float16* __restrict__ ckwT,
                                               _Float16* __restrict__ cvwT) {
  int dd = blockIdx.x, g = blockIdx.y;
  for (int f = threadIdx.x; f < 2048; f += 256) {
    _Float16 a = (_Float16)0.f, b = (_Float16)0.f;
    if (dd < D_) {
      a = (_Float16)CKW[((size_t)g * 2048 + f) * D_ + dd];
      b = (_Float16)CVW[((size_t)g * 2048 + f) * D_ + dd];
    }
    ckwT[((size_t)g * 128 + dd) * 2048 + f] = a;
    cvwT[((size_t)g * 128 + dd) * 2048 + f] = b;
  }
}

// ---------------- ck/cv f32 (G,128,128) -> ckb (G,128,64) f16 and cvT (G,64,128) f16 ----------------

__global__ void prep_ckcv(const float* __restrict__ ckf, const float* __restrict__ cvf,
                          _Float16* __restrict__ ckb, _Float16* __restrict__ cvTb) {
  int c = blockIdx.x, g = blockIdx.y, dd = threadIdx.x;  // 64 threads
  float a = ckf[((size_t)g * 128 + c) * 128 + dd];
  float b = cvf[((size_t)g * 128 + c) * 128 + dd];
  ckb[((size_t)g * 128 + c) * 64 + dd]  = (_Float16)a;
  cvTb[((size_t)g * 64 + dd) * 128 + c] = (_Float16)b;
}

// ---------------- fused compressed attention: scores, softmax, block-scores, P@cv ----------------
// block: 64 rows = 4 t x 16 h for one g. 8 waves: 2(m) x 4(n).

__global__ __launch_bounds__(256) void comp_attn(const _Float16* __restrict__ qab,
                                                 const _Float16* __restrict__ ckb,
                                                 const _Float16* __restrict__ cvTb,
                                                 float* __restrict__ comp_out,
                                                 float* __restrict__ score) {
  __shared__ float    Sbuf[64 * 128];
  __shared__ _Float16 Pbuf[64 * 128];
  const int t0 = blockIdx.x * 4;
  const int g  = blockIdx.y;
  const int tid = threadIdx.x, lane = tid & 31, wave = tid >> 5;
  const int wm = wave >> 2, wn = wave & 3;
  const float scale = 0.125f;  // 1/sqrt(64)
  const _Float16* qbase = qab + ((size_t)g * (T_ * 16) + (size_t)t0 * 16) * D_;
  const _Float16* ckg   = ckb + (size_t)g * 128 * 64;
  const int mo = (lane >> 4) * 8, nn = lane & 15;

  // phase 1: S(64x128) = Q * ck^T
  v8f zacc = {};
  v8f s[2][2];
  s[0][0] = s[0][1] = s[1][0] = s[1][1] = zacc;
  #pragma unroll
  for (int kk = 0; kk < 64; kk += 32) {
    v16h af[2], bf[2];
    #pragma unroll
    for (int mi = 0; mi < 2; ++mi) af[mi] = load_a_frag(qbase + (wm * 32 + mi * 16) * D_ + kk, D_, lane);
    #pragma unroll
    for (int ni = 0; ni < 2; ++ni) bf[ni] = load_b_frag(ckg + (wn * 32 + ni * 16) * 64 + kk, 64, lane);
    #pragma unroll
    for (int mi = 0; mi < 2; ++mi)
      #pragma unroll
      for (int ni = 0; ni < 2; ++ni)
        s[mi][ni] = wmma32(af[mi], bf[ni], s[mi][ni]);
  }
  #pragma unroll
  for (int mi = 0; mi < 2; ++mi)
    #pragma unroll
    for (int ni = 0; ni < 2; ++ni)
      #pragma unroll
      for (int r = 0; r < 8; ++r)
        Sbuf[(wm * 32 + mi * 16 + mo + r) * 128 + wn * 32 + ni * 16 + nn] = s[mi][ni][r] * scale;
  __syncthreads();

  // phase 2: masked softmax per row
  if (tid < 64) {
    int r = tid, t = t0 + (r >> 4);
    int nv = (t >= 31) ? (((t - 31) >> 4) + 1) : 0;
    if (nv > NCHUNK) nv = NCHUNK;
    float* row = Sbuf + r * 128;
    if (nv == 0) {
      for (int c = 0; c < 128; ++c) Pbuf[r * 128 + c] = (_Float16)0.f;
    } else {
      float mx = -3e38f;
      for (int c = 0; c < nv; ++c) mx = fmaxf(mx, row[c]);
      float sum = 0.f;
      for (int c = 0; c < 128; ++c) {
        float e = (c < nv) ? __expf(row[c] - mx) : 0.f;
        row[c] = e; sum += e;
      }
      float inv = 1.f / sum;
      for (int c = 0; c < 128; ++c) Pbuf[r * 128 + c] = (_Float16)(row[c] * inv);
    }
  }
  __syncthreads();

  // phase 3: block scores (sum over h, chunk->block overlap = 5 chunks)
  if (tid < 128) {
    int tl = tid >> 5, b = tid & 31;
    int clo = 4 * b - 1; if (clo < 0) clo = 0;
    int chi = 4 * b + 3; if (chi > NCHUNK - 1) chi = NCHUNK - 1;
    float sacc = 0.f;
    for (int h = 0; h < 16; ++h)
      for (int c = clo; c <= chi; ++c)
        sacc += (float)Pbuf[(tl * 16 + h) * 128 + c];
    score[((size_t)g * T_ + t0 + tl) * 32 + b] = sacc;
  }

  // phase 4: out(64x64) = P(64x128) * cvT^T
  v8f o[2];
  o[0] = o[1] = zacc;
  #pragma unroll
  for (int kk = 0; kk < 128; kk += 32) {
    v16h bfr = load_b_frag(cvTb + (size_t)g * 64 * 128 + (wn * 16) * 128 + kk, 128, lane);
    #pragma unroll
    for (int mi = 0; mi < 2; ++mi)
      o[mi] = wmma32(load_a_frag(Pbuf + (wm * 32 + mi * 16) * 128 + kk, 128, lane), bfr, o[mi]);
  }
  #pragma unroll
  for (int mi = 0; mi < 2; ++mi)
    #pragma unroll
    for (int r = 0; r < 8; ++r) {
      int rt = wm * 32 + mi * 16 + mo + r;
      int t = t0 + (rt >> 4), h = rt & 15;
      comp_out[((size_t)t * HQ_ + g * 16 + h) * D_ + wn * 16 + nn] = o[mi][r];
    }
}

// ---------------- top-k block selection -> 32-bit mask per (g,t) ----------------

__global__ void topk_kernel(const float* __restrict__ score, unsigned* __restrict__ selmask) {
  int idx = blockIdx.x * blockDim.x + threadIdx.x;
  if (idx >= G_ * T_) return;
  int t = idx & (T_ - 1);
  int qb = t >> 6;
  float v[32];
  const float* srow = score + (size_t)idx * 32;
  for (int b = 0; b < 32; ++b) {
    bool causal = b <= qb;
    bool forced = (b < 1) || (qb - b < 2);
    v[b] = causal ? (forced ? 3.0e38f : srow[b]) : -3.0e38f;
  }
  unsigned mask = 0;
  for (int it = 0; it < TOPK_; ++it) {
    float bv = -3.0e38f; int bi = -1;
    for (int b = 0; b < 32; ++b)
      if (!((mask >> b) & 1) && v[b] > bv) { bv = v[b]; bi = b; }
    if (bi >= 0) mask |= 1u << bi;
  }
  selmask[idx] = mask;
}

// ---------------- fused block-sparse + sliding-window flash attention ----------------
// block: 64 rows = 4 t x 16 h for one g; scan causal 64-key blocks; two online-softmax states.

__global__ __launch_bounds__(256) void fused_attn(const _Float16* __restrict__ qab,
                                                  const _Float16* __restrict__ kb,
                                                  const _Float16* __restrict__ vb,
                                                  const unsigned* __restrict__ selmask,
                                                  float* __restrict__ sp_out,
                                                  float* __restrict__ sl_out) {
  __shared__ float    Sbuf[64 * 64];
  __shared__ _Float16 Psp[64 * 64];
  __shared__ _Float16 Psl[64 * 64];
  __shared__ _Float16 Vt[64 * 64];
  __shared__ float msp[64], lsp[64], csp[64];
  __shared__ float msl[64], lsl[64], csl[64];
  __shared__ unsigned selv[4];
  const int t0 = blockIdx.x * 4;
  const int g  = blockIdx.y;
  const int tid = threadIdx.x, lane = tid & 31, wave = tid >> 5;
  const int wm = wave >> 2, wn = wave & 3;
  const float scale = 0.125f;
  const int mo = (lane >> 4) * 8, nn = lane & 15;

  if (tid < 64) { msp[tid] = -3e38f; lsp[tid] = 0.f; msl[tid] = -3e38f; lsl[tid] = 0.f; }
  if (tid < 4) selv[tid] = selmask[(size_t)g * T_ + t0 + tid];
  __syncthreads();

  const _Float16* qbase = qab + ((size_t)g * (T_ * 16) + (size_t)t0 * 16) * D_;
  v8f zacc = {};
  v8f asp[2], asl[2];
  asp[0] = asp[1] = asl[0] = asl[1] = zacc;
  const unsigned selall = selv[0] | selv[1] | selv[2] | selv[3];
  const int bmax = (t0 + 3) >> 6;

  for (int b = 0; b <= bmax; ++b) {
    const int s0 = b * 64;
    const bool sparse_any = ((selall >> b) & 1) != 0;
    const bool slide_any  = (s0 + 63) >= (t0 - WINDOW_);
    if (!sparse_any && !slide_any) continue;

    // stage V block transposed: Vt[d][s]
    for (int i = tid; i < 4096; i += 256) {
      int si = i >> 6, d = i & 63;
      Vt[d * 64 + si] = vb[((size_t)(s0 + si) * G_ + g) * D_ + d];
    }
    // S = Q * K^T (64 rows x 64 keys)
    v8f st[2];
    st[0] = st[1] = zacc;
    #pragma unroll
    for (int kk = 0; kk < 64; kk += 32) {
      v16h bfr = load_b_frag(kb + ((size_t)(s0 + wn * 16) * G_ + g) * D_ + kk, G_ * D_, lane);
      #pragma unroll
      for (int mi = 0; mi < 2; ++mi)
        st[mi] = wmma32(load_a_frag(qbase + (wm * 32 + mi * 16) * D_ + kk, D_, lane), bfr, st[mi]);
    }
    #pragma unroll
    for (int mi = 0; mi < 2; ++mi)
      #pragma unroll
      for (int r = 0; r < 8; ++r)
        Sbuf[(wm * 32 + mi * 16 + mo + r) * 64 + wn * 16 + nn] = st[mi][r] * scale;
    __syncthreads();

    // online softmax update per row
    if (tid < 64) {
      const int r = tid;
      const int t = t0 + (r >> 4);
      const float* row = Sbuf + r * 64;
      { // sparse branch
        float corr = 1.f;
        if ((selv[r >> 4] >> b) & 1) {
          float mx = -3e38f;
          for (int j = 0; j < 64; ++j) if (s0 + j <= t) mx = fmaxf(mx, row[j]);
          float mnew = fmaxf(msp[r], mx);
          corr = __expf(msp[r] - mnew);
          float sum = 0.f;
          for (int j = 0; j < 64; ++j) {
            float e = (s0 + j <= t) ? __expf(row[j] - mnew) : 0.f;
            Psp[r * 64 + j] = (_Float16)e; sum += e;
          }
          lsp[r] = lsp[r] * corr + sum;
          msp[r] = mnew;
        } else {
          for (int j = 0; j < 64; ++j) Psp[r * 64 + j] = (_Float16)0.f;
        }
        csp[r] = corr;
      }
      { // sliding branch
        float corr = 1.f;
        float mx = -3e38f;
        for (int j = 0; j < 64; ++j) {
          int sidx = s0 + j;
          if (sidx <= t && t - sidx <= WINDOW_) mx = fmaxf(mx, row[j]);
        }
        if (mx > -2e38f) {
          float mnew = fmaxf(msl[r], mx);
          corr = __expf(msl[r] - mnew);
          float sum = 0.f;
          for (int j = 0; j < 64; ++j) {
            int sidx = s0 + j;
            float e = (sidx <= t && t - sidx <= WINDOW_) ? __expf(row[j] - mnew) : 0.f;
            Psl[r * 64 + j] = (_Float16)e; sum += e;
          }
          lsl[r] = lsl[r] * corr + sum;
          msl[r] = mnew;
        } else {
          for (int j = 0; j < 64; ++j) Psl[r * 64 + j] = (_Float16)0.f;
        }
        csl[r] = corr;
      }
    }
    __syncthreads();

    // rescale accumulators, then acc += P @ V
    #pragma unroll
    for (int mi = 0; mi < 2; ++mi) {
      const int rowb = wm * 32 + mi * 16;
      #pragma unroll
      for (int r = 0; r < 8; ++r) {
        asp[mi][r] *= csp[rowb + mo + r];
        asl[mi][r] *= csl[rowb + mo + r];
      }
    }
    #pragma unroll
    for (int kk = 0; kk < 64; kk += 32) {
      v16h bfr = load_b_frag(Vt + (wn * 16) * 64 + kk, 64, lane);
      #pragma unroll
      for (int mi = 0; mi < 2; ++mi) {
        const int rowb = wm * 32 + mi * 16;
        asp[mi] = wmma32(load_a_frag(Psp + rowb * 64 + kk, 64, lane), bfr, asp[mi]);
        asl[mi] = wmma32(load_a_frag(Psl + rowb * 64 + kk, 64, lane), bfr, asl[mi]);
      }
    }
    __syncthreads();
  }

  // normalize + store
  #pragma unroll
  for (int mi = 0; mi < 2; ++mi) {
    const int rowb = wm * 32 + mi * 16;
    #pragma unroll
    for (int r = 0; r < 8; ++r) {
      int rt = rowb + mo + r;
      int t = t0 + (rt >> 4), h = rt & 15;
      float l1 = lsp[rt], l2 = lsl[rt];
      size_t idx = ((size_t)t * HQ_ + g * 16 + h) * D_ + wn * 16 + nn;
      sp_out[idx] = (l1 > 0.f) ? asp[mi][r] / l1 : 0.f;
      sl_out[idx] = (l2 > 0.f) ? asl[mi][r] / l2 : 0.f;
    }
  }
}

// ---------------- gated combine -> f16 for final GEMM ----------------

__global__ void combine_kernel(const float* __restrict__ gate, const float* __restrict__ comp,
                               const float* __restrict__ sp, const float* __restrict__ sl,
                               _Float16* __restrict__ outb) {
  size_t stride = (size_t)gridDim.x * blockDim.x;
  for (size_t i = (size_t)blockIdx.x * blockDim.x + threadIdx.x; i < (size_t)T_ * HQ_ * D_; i += stride) {
    int t = (int)(i >> 11);  // HQ_*D_ = 2048
    outb[i] = (_Float16)(gate[t * 3] * comp[i] + gate[t * 3 + 1] * sp[i] + gate[t * 3 + 2] * sl[i]);
  }
}

// ---------------- host orchestration ----------------

extern "C" void kernel_launch(void* const* d_in, const int* in_sizes, int n_in,
                              void* d_out, int out_size, void* d_ws, size_t ws_size,
                              hipStream_t stream) {
  const float* x   = (const float*)d_in[0];
  const float* Wq  = (const float*)d_in[1];
  const float* Wk  = (const float*)d_in[2];
  const float* Wv  = (const float*)d_in[3];
  const float* Wo  = (const float*)d_in[4];
  const float* Wg  = (const float*)d_in[5];
  const float* CKW = (const float*)d_in[6];
  const float* CVW = (const float*)d_in[7];

  char* p = (char*)d_ws;
  auto alloc = [&](size_t bytes) -> void* {
    void* r = (void*)p;
    p += (bytes + 255) & ~(size_t)255;
    return r;
  };
  _Float16* xb   = (_Float16*)alloc((size_t)T_ * HID * 2);
  _Float16* Wqb  = (_Float16*)alloc((size_t)HQ_ * D_ * HID * 2);
  _Float16* Wkb  = (_Float16*)alloc((size_t)G_ * D_ * HID * 2);
  _Float16* Wvb  = (_Float16*)alloc((size_t)G_ * D_ * HID * 2);
  _Float16* Wob  = (_Float16*)alloc((size_t)HID * HQ_ * D_ * 2);
  float*    qf   = (float*)alloc((size_t)T_ * HQ_ * D_ * 4);
  float*    kf   = (float*)alloc((size_t)T_ * G_ * D_ * 4);
  float*    vf   = (float*)alloc((size_t)T_ * G_ * D_ * 4);
  float*    gate = (float*)alloc((size_t)T_ * 3 * 4);
  _Float16* qab  = (_Float16*)alloc((size_t)G_ * T_ * 16 * D_ * 2);
  _Float16* kbb  = (_Float16*)alloc((size_t)T_ * G_ * D_ * 2);
  _Float16* vbb  = (_Float16*)alloc((size_t)T_ * G_ * D_ * 2);
  _Float16* kwb  = (_Float16*)alloc((size_t)G_ * 128 * 2048 * 2);
  _Float16* vwb  = (_Float16*)alloc((size_t)G_ * 128 * 2048 * 2);
  _Float16* ckwT = (_Float16*)alloc((size_t)G_ * 128 * 2048 * 2);
  _Float16* cvwT = (_Float16*)alloc((size_t)G_ * 128 * 2048 * 2);
  float*    ckf  = (float*)alloc((size_t)G_ * 128 * 128 * 4);
  float*    cvf  = (float*)alloc((size_t)G_ * 128 * 128 * 4);
  _Float16* ckb  = (_Float16*)alloc((size_t)G_ * 128 * 64 * 2);
  _Float16* cvTb = (_Float16*)alloc((size_t)G_ * 64 * 128 * 2);
  float*    comp = (float*)alloc((size_t)T_ * HQ_ * D_ * 4);
  float*    scor = (float*)alloc((size_t)G_ * T_ * 32 * 4);
  unsigned* selm = (unsigned*)alloc((size_t)G_ * T_ * 4);
  float*    spo  = (float*)alloc((size_t)T_ * HQ_ * D_ * 4);
  float*    slo  = (float*)alloc((size_t)T_ * HQ_ * D_ * 4);
  _Float16* outb = (_Float16*)alloc((size_t)T_ * HQ_ * D_ * 2);

  // 1. convert activations / weights to f16
  f32_to_f16_kernel<<<1024, 256, 0, stream>>>(x,  xb,  (size_t)T_ * HID);
  f32_to_f16_kernel<<<1024, 256, 0, stream>>>(Wq, Wqb, (size_t)HQ_ * D_ * HID);
  f32_to_f16_kernel<<<256,  256, 0, stream>>>(Wk, Wkb, (size_t)G_ * D_ * HID);
  f32_to_f16_kernel<<<256,  256, 0, stream>>>(Wv, Wvb, (size_t)G_ * D_ * HID);
  f32_to_f16_kernel<<<1024, 256, 0, stream>>>(Wo, Wob, (size_t)HID * HQ_ * D_);
  prep_cw<<<dim3(128, G_), 256, 0, stream>>>(CKW, CVW, ckwT, cvwT);

  // 2. projections (WMMA GEMMs, TDM-staged tiles)
  gemm_f16<<<dim3(16, 16, 1), 256, 0, stream>>>(xb, Wqb, qf, HID, HQ_ * D_, 0, 0, 0);
  gemm_f16<<<dim3(1, 16, 1),  256, 0, stream>>>(xb, Wkb, kf, HID, G_ * D_, 0, 0, 0);
  gemm_f16<<<dim3(1, 16, 1),  256, 0, stream>>>(xb, Wvb, vf, HID, G_ * D_, 0, 0, 0);
  gate_kernel<<<T_, 256, 0, stream>>>(x, Wg, gate);

  // 3. RoPE + repack, build compression windows
  rope_pack<<<T_, 256, 0, stream>>>(qf, kf, vf, qab, kbb, vbb);
  build_windows<<<dim3(128, G_), 256, 0, stream>>>(kbb, vbb, kwb, vwb);

  // 4. compressed K/V (batched over G): [128 x 2048] @ [128 x 2048]^T
  gemm_f16<<<dim3(1, 1, G_), 256, 0, stream>>>(kwb, ckwT, ckf, 2048, 128,
                                               (size_t)128 * 2048, (size_t)128 * 2048, (size_t)128 * 128);
  gemm_f16<<<dim3(1, 1, G_), 256, 0, stream>>>(vwb, cvwT, cvf, 2048, 128,
                                               (size_t)128 * 2048, (size_t)128 * 2048, (size_t)128 * 128);
  prep_ckcv<<<dim3(128, G_), 64, 0, stream>>>(ckf, cvf, ckb, cvTb);

  // 5. compressed attention + block scores, then top-k masks
  comp_attn<<<dim3(T_ / 4, G_), 256, 0, stream>>>(qab, ckb, cvTb, comp, scor);
  topk_kernel<<<(G_ * T_ + 255) / 256, 256, 0, stream>>>(scor, selm);

  // 6. fused block-sparse + sliding-window flash attention
  fused_attn<<<dim3(T_ / 4, G_), 256, 0, stream>>>(qab, kbb, vbb, selm, spo, slo);

  // 7. gated combine + output projection
  combine_kernel<<<2048, 256, 0, stream>>>(gate, comp, spo, slo, outb);
  gemm_f16<<<dim3(16, 16, 1), 256, 0, stream>>>(outb, Wob, (float*)d_out, HID, HID, 0, 0, 0);

  (void)in_sizes; (void)n_in; (void)out_size; (void)ws_size;
}
